// InstanceRelationModule_19713899889181
// MI455X (gfx1250) — matrix operand order repaired and checked by reading, not compiled, not run on hardware
//
#include <hip/hip_runtime.h>

// ---------------------------------------------------------------------------
// InstanceRelationModule for MI455X (gfx1250, wave32, WMMA bf16 -> f32 acc)
//
// rel[c,i,j] = (X_c[i,:]·X_c[j,:])/96 + (param_i·param_j)/480
// att        = softmax_c(rel) * same_img_mask
// out        = relu(att_c @ X_c + feats)   in [P,C,H,W] layout
//
// Fast path (ws >= 117.4MB): bf16 copy of X + bf16 att in d_ws, async
// global->LDS staging (double-buffered in k_rel), symmetric rel (lower
// triangle only, mirrored stores). f32 rel scratch lives in d_out (fully
// overwritten by k_out afterwards; stream-ordered so this is safe).
// Fallback (ws >= 33.5MB): round-1 style f32 staging with in-flight cvt.
// ---------------------------------------------------------------------------

#define PN     512
#define CN     32
#define HWN    3072
#define PDIM   34
#define INV_NORM   (1.0f / 96.0f)
#define INV_PNORM  (1.0f / 480.0f)

#define LDS_S1 33   // dword row stride, 64-bf16 tiles (conflict-free frags)
#define LDS_S3 35   // dword row stride, transposed B tile in k_out

typedef __attribute__((ext_vector_type(16))) __bf16 v16bf;
typedef __attribute__((ext_vector_type(8)))  float  v8f;
union FragU { unsigned u[8]; v16bf v; };

static __device__ __forceinline__ unsigned short f2bf(float f) {
    unsigned u = __float_as_uint(f);
    u += 0x7FFFu + ((u >> 16) & 1u);       // RNE, finite inputs
    return (unsigned short)(u >> 16);
}
static __device__ __forceinline__ unsigned pack_bf16(float a, float b) {
    return (unsigned)f2bf(a) | ((unsigned)f2bf(b) << 16);
}
// A-matrix (16x32 bf16, ISA 7.12.2): lane l = (n=l%16, h=l/16); VGPR v packs
// K = {2v+8h,+1} (v<4) | {8+2v+8h,+1} (v>=4)  -> dword col v+4h | v+4+4h
static __device__ __forceinline__ int a_col(int v, int h) {
    return (v < 4) ? (v + 4 * h) : (v + 4 + 4 * h);
}
// B as loaded here (contiguous k, as A in the kernels): VGPR v packs K = {2v+16h,+1} -> dword col v+8h

// --- CDNA5 async global->LDS (ISA 15.18.3 op 96) + ASYNCcnt wait (SOPP 74) --
static __device__ __forceinline__ void async_b32(unsigned lds_off, const void* g) {
    asm volatile("global_load_async_to_lds_b32 %0, %1, off"
                 :: "v"(lds_off), "v"(g) : "memory");
}
static __device__ __forceinline__ void wait_async0() {
    asm volatile("s_wait_asynccnt 0x0" ::: "memory");
}
// LDS byte offset of a __shared__ object: flat addr low 32 bits (ISA 10.2)
static __device__ __forceinline__ unsigned lds_off_of(const void* p) {
    return (unsigned)(unsigned long long)p;
}

// ===========================================================================
// k_cvt: feats f32 [P,C,HW] -> packed bf16 copy (8 elements / thread)
// ===========================================================================
__global__ __launch_bounds__(256)
void k_cvt(const float* __restrict__ in, unsigned* __restrict__ outw) {
    const size_t t = (size_t)blockIdx.x * 256 + threadIdx.x;
    const float4* in4 = (const float4*)in;
    float4 a = in4[2 * t], b = in4[2 * t + 1];
    uint4 r;
    r.x = pack_bf16(a.x, a.y); r.y = pack_bf16(a.z, a.w);
    r.z = pack_bf16(b.x, b.y); r.w = pack_bf16(b.z, b.w);
    ((uint4*)outw)[t] = r;
}

// ===========================================================================
// k_rel_bf: rel_raw[c][i][j] = X_c[i,:]·X_c[j,:]  (bf16 in, f32 out)
// lower-triangle blocks only; async double-buffered LDS staging.
// ===========================================================================
__global__ __launch_bounds__(256)
void k_rel_bf(const unsigned* __restrict__ Xbf, float* __restrict__ rel) {
    const int c  = blockIdx.z;
    const int i0 = blockIdx.y * 64;
    const int j0 = blockIdx.x * 64;
    if (j0 > i0) return;                    // symmetry: mirror handles it

    __shared__ unsigned As[2][64 * LDS_S1];
    __shared__ unsigned Bs[2][64 * LDS_S1];

    const int tid  = threadIdx.x;
    const int lane = tid & 31;
    const int n    = lane & 15;
    const int h    = lane >> 4;
    const int wave = tid >> 5;
    const int isub = (wave & 3) * 16;
    const int jsub = (wave >> 2) * 32;

    const unsigned ldsA = lds_off_of(&As[0][0]);
    const unsigned ldsB = lds_off_of(&Bs[0][0]);
    const size_t rstride = (size_t)CN * (HWN / 2);          // dwords / instance
    const unsigned* gA0 = Xbf + (size_t)i0 * rstride + (size_t)c * (HWN / 2);
    const unsigned* gB0 = Xbf + (size_t)j0 * rstride + (size_t)c * (HWN / 2);

    auto issue = [&](int kc, int p) {
        const unsigned kd = (unsigned)(kc * 32);            // k0 in dwords
        const unsigned pb = (unsigned)p * (unsigned)sizeof(As[0]);
#pragma unroll
        for (int q = 0; q < 8; ++q) {
            const int f   = tid + 256 * q;                  // one row / wave
            const int row = f >> 5, col = f & 31;
            const unsigned loff = (unsigned)((row * LDS_S1 + col) * 4);
            const size_t   goff = (size_t)row * rstride + kd + col;
            async_b32(ldsA + pb + loff, gA0 + goff);
            async_b32(ldsB + pb + loff, gB0 + goff);
        }
    };

    v8f acc0 = {0.f,0.f,0.f,0.f,0.f,0.f,0.f,0.f};
    v8f acc1 = {0.f,0.f,0.f,0.f,0.f,0.f,0.f,0.f};
    const int arow  = (isub + n) * LDS_S1;
    const int brow0 = (jsub + n) * LDS_S1;
    const int brow1 = (jsub + 16 + n) * LDS_S1;

    const int NCHUNK = HWN / 64;                            // 48
    issue(0, 0);
    for (int kc = 0; kc < NCHUNK; ++kc) {
        const int p = kc & 1;
        wait_async0();
        __syncthreads();
        if (kc + 1 < NCHUNK) issue(kc + 1, p ^ 1);          // overlap w/ WMMA
        const unsigned* A = As[p];
        const unsigned* B = Bs[p];
#pragma unroll
        for (int ks = 0; ks < 2; ++ks) {
            const int kp = ks * 16;
            FragU fa, fb0, fb1;
#pragma unroll
            for (int v = 0; v < 8; ++v) {
                fa.u[v]  = A[arow  + kp + v + 8 * h];
                fb0.u[v] = B[brow0 + kp + v + 8 * h];
                fb1.u[v] = B[brow1 + kp + v + 8 * h];
            }
            acc0 = __builtin_amdgcn_wmma_f32_16x16x32_bf16(
                       false, fa.v, false, fb0.v, (short)0, acc0, false, false);
            acc1 = __builtin_amdgcn_wmma_f32_16x16x32_bf16(
                       false, fa.v, false, fb1.v, (short)0, acc1, false, false);
        }
    }

    float* r0 = rel + (size_t)c * PN * PN;
#pragma unroll
    for (int r = 0; r < 8; ++r) {
        const int i  = i0 + isub + r + 8 * h;
        const int j1 = j0 + jsub + n;
        r0[(size_t)i * PN + j1]      = acc0[r];
        r0[(size_t)i * PN + j1 + 16] = acc1[r];
    }
    if (i0 != j0) {                                         // mirror block
#pragma unroll
        for (int r = 0; r < 8; ++r) {
            const int i  = i0 + isub + r + 8 * h;
            const int j1 = j0 + jsub + n;
            r0[(size_t)j1 * PN + i]        = acc0[r];
            r0[(size_t)(j1 + 16) * PN + i] = acc1[r];
        }
    }
}

// ===========================================================================
// k_softmax_bf: softmax over channels + param bias + mask; bf16 att out
// ===========================================================================
__global__ __launch_bounds__(256)
void k_softmax_bf(const float* __restrict__ rel, const int* __restrict__ imgid,
                  const float* __restrict__ param,
                  unsigned short* __restrict__ attbf) {
    const int flat = blockIdx.x * 256 + threadIdx.x;
    const int i = flat >> 9;
    const int j = flat & (PN - 1);

    const float* pi = param + i * PDIM;
    const float* pj = param + j * PDIM;
    float pd = 0.f;
#pragma unroll
    for (int t = 0; t < PDIM; ++t) pd += pi[t] * pj[t];
    pd *= INV_PNORM;

    const size_t base = (size_t)i * PN + j;
    float v[CN], m = -1e30f;
#pragma unroll
    for (int cc = 0; cc < CN; ++cc) {
        float t = rel[(size_t)cc * PN * PN + base] * INV_NORM + pd;
        v[cc] = t; m = fmaxf(m, t);
    }
    float s = 0.f;
#pragma unroll
    for (int cc = 0; cc < CN; ++cc) { v[cc] = __expf(v[cc] - m); s += v[cc]; }
    const float mask = (imgid[i] == imgid[j]) ? 1.f : 0.f;
    const float inv  = mask / s;
#pragma unroll
    for (int cc = 0; cc < CN; ++cc)
        attbf[(size_t)cc * PN * PN + base] = f2bf(v[cc] * inv);
}

// ===========================================================================
// k_out_bf: out = relu(att_c @ X_c + feats); A tile async, B tile transposed
// ===========================================================================
__global__ __launch_bounds__(256)
void k_out_bf(const float* __restrict__ feats, const unsigned* __restrict__ Xbf,
              const unsigned short* __restrict__ attbf, float* __restrict__ out) {
    const int c  = blockIdx.z;
    const int i0 = blockIdx.y * 64;
    const int d0 = blockIdx.x * 64;

    __shared__ unsigned As[64 * LDS_S1];
    __shared__ unsigned Bs[64 * LDS_S3];
    unsigned short* Bs16 = (unsigned short*)Bs;

    const int tid  = threadIdx.x;
    const int lane = tid & 31;
    const int n    = lane & 15;
    const int h    = lane >> 4;
    const int wave = tid >> 5;
    const int isub = (wave & 3) * 16;
    const int dsub = (wave >> 2) * 32;
    const int srow = tid >> 2;
    const int sd   = (tid & 3) * 16;

    const unsigned ldsA = lds_off_of(&As[0]);
    const unsigned* attU = (const unsigned*)attbf;
    const unsigned* gA0  = attU + ((size_t)c * PN * PN + (size_t)i0 * PN) / 2;

    v8f acc0 = {0.f,0.f,0.f,0.f,0.f,0.f,0.f,0.f};
    v8f acc1 = {0.f,0.f,0.f,0.f,0.f,0.f,0.f,0.f};
    const int arow  = (isub + n) * LDS_S1;
    const int brow0 = (dsub + n) * LDS_S3;
    const int brow1 = (dsub + 16 + n) * LDS_S3;

    for (int k0 = 0; k0 < PN; k0 += 64) {
        __syncthreads();
        // A tile (64 i-rows x 32 dwords of att) via async DMA
#pragma unroll
        for (int q = 0; q < 8; ++q) {
            const int f   = tid + 256 * q;
            const int row = f >> 5, col = f & 31;
            async_b32(ldsA + (unsigned)((row * LDS_S1 + col) * 4),
                      gA0 + (size_t)row * (PN / 2) + (k0 >> 1) + col);
        }
        // B tile: X_c rows k, transposed to [d][k] (16-bit scatter)
        {
            const uint4* gb = (const uint4*)(Xbf +
                (((size_t)(k0 + srow) * CN + c) * HWN + d0 + sd) / 2);
            uint4 b0 = gb[0], b1 = gb[1];
            unsigned w[8] = {b0.x, b0.y, b0.z, b0.w, b1.x, b1.y, b1.z, b1.w};
#pragma unroll
            for (int e = 0; e < 8; ++e) {
                Bs16[(sd + 2*e + 0) * (2 * LDS_S3) + srow] =
                    (unsigned short)(w[e] & 0xFFFFu);
                Bs16[(sd + 2*e + 1) * (2 * LDS_S3) + srow] =
                    (unsigned short)(w[e] >> 16);
            }
        }
        wait_async0();
        __syncthreads();
#pragma unroll
        for (int ks = 0; ks < 2; ++ks) {
            const int kp = ks * 16;
            FragU fa, fb0, fb1;
#pragma unroll
            for (int v = 0; v < 8; ++v) {
                fa.u[v]  = As[arow  + kp + v + 8 * h];
                fb0.u[v] = Bs[brow0 + kp + v + 8 * h];
                fb1.u[v] = Bs[brow1 + kp + v + 8 * h];
            }
            acc0 = __builtin_amdgcn_wmma_f32_16x16x32_bf16(
                       false, fa.v, false, fb0.v, (short)0, acc0, false, false);
            acc1 = __builtin_amdgcn_wmma_f32_16x16x32_bf16(
                       false, fa.v, false, fb1.v, (short)0, acc1, false, false);
        }
    }

#pragma unroll
    for (int r = 0; r < 8; ++r) {
        const int i = i0 + isub + r + 8 * h;
        const size_t g = ((size_t)i * CN + c) * HWN + d0 + dsub + n;
        out[g]      = fmaxf(acc0[r] + feats[g],      0.f);
        out[g + 16] = fmaxf(acc1[r] + feats[g + 16], 0.f);
    }
}

// ===========================================================================
// Fallback path (small d_ws): f32-staged kernels (round-1 style + symmetry)
// ===========================================================================
__global__ __launch_bounds__(256)
void k_rel_f32(const float* __restrict__ feats, float* __restrict__ rel) {
    const int c  = blockIdx.z;
    const int i0 = blockIdx.y * 64;
    const int j0 = blockIdx.x * 64;
    if (j0 > i0) return;

    __shared__ unsigned As[64 * LDS_S1];
    __shared__ unsigned Bs[64 * LDS_S1];

    const int tid  = threadIdx.x;
    const int lane = tid & 31;
    const int n    = lane & 15;
    const int h    = lane >> 4;
    const int wave = tid >> 5;
    const int isub = (wave & 3) * 16;
    const int jsub = (wave >> 2) * 32;
    const int srow = tid >> 2;
    const int scol = (tid & 3) * 16;

    v8f acc0 = {0.f,0.f,0.f,0.f,0.f,0.f,0.f,0.f};
    v8f acc1 = {0.f,0.f,0.f,0.f,0.f,0.f,0.f,0.f};
    const int arow  = (isub + n) * LDS_S1;
    const int brow0 = (jsub + n) * LDS_S1;
    const int brow1 = (jsub + 16 + n) * LDS_S1;

    for (int k0 = 0; k0 < HWN; k0 += 64) {
        __syncthreads();
        {
            const float4* ga = (const float4*)(feats +
                (size_t)((i0 + srow) * CN + c) * HWN + k0 + scol);
            const float4* gb = (const float4*)(feats +
                (size_t)((j0 + srow) * CN + c) * HWN + k0 + scol);
            const int base = srow * LDS_S1 + (scol >> 1);
#pragma unroll
            for (int q = 0; q < 4; ++q) {
                float4 a = ga[q], b = gb[q];
                As[base + 2*q + 0] = pack_bf16(a.x, a.y);
                As[base + 2*q + 1] = pack_bf16(a.z, a.w);
                Bs[base + 2*q + 0] = pack_bf16(b.x, b.y);
                Bs[base + 2*q + 1] = pack_bf16(b.z, b.w);
            }
        }
        __syncthreads();
#pragma unroll
        for (int ks = 0; ks < 2; ++ks) {
            const int kp = ks * 16;
            FragU fa, fb0, fb1;
#pragma unroll
            for (int v = 0; v < 8; ++v) {
                fa.u[v]  = As[arow  + kp + v + 8 * h];
                fb0.u[v] = Bs[brow0 + kp + v + 8 * h];
                fb1.u[v] = Bs[brow1 + kp + v + 8 * h];
            }
            acc0 = __builtin_amdgcn_wmma_f32_16x16x32_bf16(
                       false, fa.v, false, fb0.v, (short)0, acc0, false, false);
            acc1 = __builtin_amdgcn_wmma_f32_16x16x32_bf16(
                       false, fa.v, false, fb1.v, (short)0, acc1, false, false);
        }
    }

    float* r0 = rel + (size_t)c * PN * PN;
#pragma unroll
    for (int r = 0; r < 8; ++r) {
        const int i  = i0 + isub + r + 8 * h;
        const int j1 = j0 + jsub + n;
        r0[(size_t)i * PN + j1]      = acc0[r];
        r0[(size_t)i * PN + j1 + 16] = acc1[r];
    }
    if (i0 != j0) {
#pragma unroll
        for (int r = 0; r < 8; ++r) {
            const int i  = i0 + isub + r + 8 * h;
            const int j1 = j0 + jsub + n;
            r0[(size_t)j1 * PN + i]        = acc0[r];
            r0[(size_t)(j1 + 16) * PN + i] = acc1[r];
        }
    }
}

__global__ __launch_bounds__(256)
void k_softmax_f32(const float* __restrict__ rel, const int* __restrict__ imgid,
                   const float* __restrict__ param, float* __restrict__ att) {
    const int flat = blockIdx.x * 256 + threadIdx.x;
    const int i = flat >> 9;
    const int j = flat & (PN - 1);
    const float* pi = param + i * PDIM;
    const float* pj = param + j * PDIM;
    float pd = 0.f;
#pragma unroll
    for (int t = 0; t < PDIM; ++t) pd += pi[t] * pj[t];
    pd *= INV_PNORM;
    const size_t base = (size_t)i * PN + j;
    float v[CN], m = -1e30f;
#pragma unroll
    for (int cc = 0; cc < CN; ++cc) {
        float t = rel[(size_t)cc * PN * PN + base] * INV_NORM + pd;
        v[cc] = t; m = fmaxf(m, t);
    }
    float s = 0.f;
#pragma unroll
    for (int cc = 0; cc < CN; ++cc) { v[cc] = __expf(v[cc] - m); s += v[cc]; }
    const float mask = (imgid[i] == imgid[j]) ? 1.f : 0.f;
    const float inv  = mask / s;
#pragma unroll
    for (int cc = 0; cc < CN; ++cc)
        att[(size_t)cc * PN * PN + base] = v[cc] * inv;
}

__global__ __launch_bounds__(256)
void k_out_f32(const float* __restrict__ feats, const float* __restrict__ att,
               float* __restrict__ out) {
    const int c  = blockIdx.z;
    const int i0 = blockIdx.y * 64;
    const int d0 = blockIdx.x * 64;

    __shared__ unsigned As[64 * LDS_S1];
    __shared__ unsigned Bs[64 * LDS_S3];
    unsigned short* Bs16 = (unsigned short*)Bs;

    const int tid  = threadIdx.x;
    const int lane = tid & 31;
    const int n    = lane & 15;
    const int h    = lane >> 4;
    const int wave = tid >> 5;
    const int isub = (wave & 3) * 16;
    const int dsub = (wave >> 2) * 32;
    const int srow = tid >> 2;
    const int scol = (tid & 3) * 16;

    v8f acc0 = {0.f,0.f,0.f,0.f,0.f,0.f,0.f,0.f};
    v8f acc1 = {0.f,0.f,0.f,0.f,0.f,0.f,0.f,0.f};
    const int arow  = (isub + n) * LDS_S1;
    const int brow0 = (dsub + n) * LDS_S3;
    const int brow1 = (dsub + 16 + n) * LDS_S3;
    const float* att_c = att + (size_t)c * PN * PN;

    for (int k0 = 0; k0 < PN; k0 += 64) {
        __syncthreads();
        {
            const float4* ga = (const float4*)(att_c +
                (size_t)(i0 + srow) * PN + k0 + scol);
            const int abase = srow * LDS_S1 + (scol >> 1);
#pragma unroll
            for (int q = 0; q < 4; ++q) {
                float4 a = ga[q];
                As[abase + 2*q + 0] = pack_bf16(a.x, a.y);
                As[abase + 2*q + 1] = pack_bf16(a.z, a.w);
            }
            const float4* gb = (const float4*)(feats +
                (size_t)((k0 + srow) * CN + c) * HWN + d0 + scol);
#pragma unroll
            for (int q = 0; q < 4; ++q) {
                float4 b = gb[q];
                const int dbase = scol + 4 * q;
                Bs16[(dbase + 0) * (2 * LDS_S3) + srow] = f2bf(b.x);
                Bs16[(dbase + 1) * (2 * LDS_S3) + srow] = f2bf(b.y);
                Bs16[(dbase + 2) * (2 * LDS_S3) + srow] = f2bf(b.z);
                Bs16[(dbase + 3) * (2 * LDS_S3) + srow] = f2bf(b.w);
            }
        }
        __syncthreads();
#pragma unroll
        for (int ks = 0; ks < 2; ++ks) {
            const int kp = ks * 16;
            FragU fa, fb0, fb1;
#pragma unroll
            for (int v = 0; v < 8; ++v) {
                fa.u[v]  = As[arow  + kp + v + 8 * h];
                fb0.u[v] = Bs[brow0 + kp + v + 8 * h];
                fb1.u[v] = Bs[brow1 + kp + v + 8 * h];
            }
            acc0 = __builtin_amdgcn_wmma_f32_16x16x32_bf16(
                       false, fa.v, false, fb0.v, (short)0, acc0, false, false);
            acc1 = __builtin_amdgcn_wmma_f32_16x16x32_bf16(
                       false, fa.v, false, fb1.v, (short)0, acc1, false, false);
        }
    }

#pragma unroll
    for (int r = 0; r < 8; ++r) {
        const int i = i0 + isub + r + 8 * h;
        const size_t g = ((size_t)i * CN + c) * HWN + d0 + dsub + n;
        out[g]      = fmaxf(acc0[r] + feats[g],      0.f);
        out[g + 16] = fmaxf(acc1[r] + feats[g + 16], 0.f);
    }
}

// ===========================================================================
extern "C" void kernel_launch(void* const* d_in, const int* in_sizes, int n_in,
                              void* d_out, int out_size, void* d_ws, size_t ws_size,
                              hipStream_t stream) {
    (void)in_sizes; (void)n_in; (void)out_size;
    const float* feats = (const float*)d_in[0];
    const int*   imgid = (const int*)d_in[1];
    const float* param = (const float*)d_in[2];
    float* out = (float*)d_out;
    float* rel = (float*)d_out;  // f32 rel scratch in d_out (overwritten by k_out)

    const size_t XBF_BYTES = (size_t)PN * CN * HWN * 2;   // 100,663,296
    const size_t ATT_BYTES = (size_t)CN * PN * PN * 2;    //  16,777,216

    if (ws_size >= XBF_BYTES + ATT_BYTES) {
        unsigned*       Xbf   = (unsigned*)d_ws;
        unsigned short* attbf = (unsigned short*)((char*)d_ws + XBF_BYTES);
        k_cvt       <<<(PN * CN * HWN) / 2048, 256, 0, stream>>>(feats, Xbf);
        k_rel_bf    <<<dim3(PN/64, PN/64, CN), 256, 0, stream>>>(Xbf, rel);
        k_softmax_bf<<<dim3((PN*PN)/256),      256, 0, stream>>>(rel, imgid, param, attbf);
        k_out_bf    <<<dim3(HWN/64, PN/64, CN),256, 0, stream>>>(feats, Xbf, attbf, out);
    } else {
        float* att = (float*)d_ws;  // needs 33.5 MB
        k_rel_f32    <<<dim3(PN/64, PN/64, CN), 256, 0, stream>>>(feats, rel);
        k_softmax_f32<<<dim3((PN*PN)/256),      256, 0, stream>>>(rel, imgid, param, att);
        k_out_f32    <<<dim3(HWN/64, PN/64, CN),256, 0, stream>>>(feats, att, out);
    }
}
